// LSTM2d_45689862095226
// MI455X (gfx1250) — matrix operand order repaired and checked by reading, not compiled, not run on hardware
//
#include <hip/hip_runtime.h>
#include <hip/hip_bf16.h>
#include <stdint.h>

// ---------------------------------------------------------------------------
// 2D-LSTM seq2seq for MI455X (gfx1250). All GEMMs via v_wmma_f32_16x16x32_f16
// (f16 operands, f32 accumulate), 32x16 output per wave (two M-tiles share one
// B fragment). Recurrent state lives in LDS; vertical grid state staged with
// gfx1250 async global->LDS loads, ordered with s_wait_storecnt/s_wait_asynccnt.
// ---------------------------------------------------------------------------

typedef __attribute__((ext_vector_type(16))) _Float16 v16h;
typedef __attribute__((ext_vector_type(8)))  _Float16 v8h;
typedef __attribute__((ext_vector_type(8)))  float    v8f;

#define S_IN   16
#define S_OUT  16
#define BATCH  32
#define EMBED  256
#define ENC    256
#define D2     256
#define VOCAB  32000

// ---------------- device helpers ----------------

__device__ __forceinline__ float sigf(float x) { return 1.0f / (1.0f + __expf(-x)); }

// A fragment: 16x32 f16 tile of row-major A (lda in elements).
// Lane L<16: row M=L, K = {0..7} U {16..23}; lane>=16: row M=L-16, K={8..15} U {24..31}.
__device__ __forceinline__ v16h load_a16x32(const _Float16* A, int lda, int k0, int lane) {
  const int row = lane & 15;
  const int kb  = k0 + ((lane >> 4) << 3);
  const _Float16* p = A + row * lda + kb;
  v8h lo = *(const v8h*)(p);
  v8h hi = *(const v8h*)(p + 16);
  v16h r;
#pragma unroll
  for (int i = 0; i < 8; ++i) { r[i] = lo[i]; r[i + 8] = hi[i]; }
  return r;
}

// B fragment for B = W^T where W is row-major (N x K, ldw elements).
// Lane L: column N=L&15 (a row of W), K = {0..7} U {16..23} (lane<16) or {8..15} U {24..31} (lane>=16).
__device__ __forceinline__ v16h load_b16x32_wT(const _Float16* W, int ldw, int k0, int lane) {
  const int n  = lane & 15;
  const int kb = k0 + ((lane >> 4) << 3);
  const _Float16* p = W + n * ldw + kb;
  v8h lo = *(const v8h*)(p);
  v8h hi = *(const v8h*)(p + 16);
  v16h r;
#pragma unroll
  for (int i = 0; i < 8; ++i) { r[i] = lo[i]; r[i + 8] = hi[i]; }
  return r;
}

// 32x16 output job: two M-tiles (A0 rows m..m+15, A1 rows m+16..m+31) sharing
// one B fragment per K-step -> 2 v_wmma per 3 fragment loads.
__device__ __forceinline__ void wmma_job_32x16(const _Float16* A0, int lda,
                                               const _Float16* Wb, int ldw,
                                               int K, int lane, v8f& acc0, v8f& acc1) {
  const _Float16* A1 = A0 + (lda << 4);
  for (int k0 = 0; k0 < K; k0 += 32) {
    __builtin_prefetch(Wb + (lane & 15) * ldw + k0 + 128, 0, 3);  // global_prefetch_b8
    v16h b  = load_b16x32_wT(Wb, ldw, k0, lane);
    v16h a0 = load_a16x32(A0, lda, k0, lane);
    v16h a1 = load_a16x32(A1, lda, k0, lane);
    acc0 = __builtin_amdgcn_wmma_f32_16x16x32_f16(false, a0, false, b, (short)0, acc0, false, false);
    acc1 = __builtin_amdgcn_wmma_f32_16x16x32_f16(false, a1, false, b, (short)0, acc1, false, false);
  }
}

// gfx1250 async global->LDS copy (16 bytes per lane) + waits.
__device__ __forceinline__ void async_g2l_b128(unsigned lds_off, const void* gptr) {
  asm volatile("global_load_async_to_lds_b128 %0, %1, off"
               :: "v"(lds_off), "v"((unsigned long long)(uintptr_t)gptr)
               : "memory");
}
__device__ __forceinline__ void wait_async0()  { asm volatile("s_wait_asynccnt 0" ::: "memory"); }
__device__ __forceinline__ void wait_store0()  { asm volatile("s_wait_storecnt 0" ::: "memory"); }

// ---------------- small utility kernels ----------------

__global__ void k_f2h(const float* __restrict__ src, _Float16* __restrict__ dst, int n) {
  for (int i = blockIdx.x * blockDim.x + threadIdx.x; i < n; i += gridDim.x * blockDim.x)
    dst[i] = (_Float16)src[i];
}

__global__ void k_zero_u32(unsigned* __restrict__ p, int n) {
  for (int i = blockIdx.x * blockDim.x + threadIdx.x; i < n; i += gridDim.x * blockDim.x)
    p[i] = 0u;
}

// Embedding gather -> f16 rows. shift_bos: row<BATCH uses token 1 (BOS),
// else token from tok[row-BATCH] (i.e. y[:-1]).
__global__ void k_embed(const int* __restrict__ tok, const float* __restrict__ table,
                        _Float16* __restrict__ out, int rows, int shift_bos) {
  const int total = rows * EMBED;
  for (int i = blockIdx.x * blockDim.x + threadIdx.x; i < total; i += gridDim.x * blockDim.x) {
    const int r = i >> 8, e = i & 255;
    int t;
    if (shift_bos) t = (r < BATCH) ? 1 : tok[r - BATCH];
    else           t = tok[r];
    out[i] = (_Float16)table[t * EMBED + e];
  }
}

// ---------------- generic WMMA GEMM:  C[M,N] = A[M,K] * W[N,K]^T (+bias) ----------------
// One wave per 32x16 output job. M mult of 32; N mult of 16; K mult of 32.
__global__ void k_gemm_xwT(const _Float16* __restrict__ A, int lda,
                           const _Float16* __restrict__ W, int ldw,
                           const float* __restrict__ bias,
                           float* __restrict__ C, int ldc,
                           int M, int N, int K) {
  const int lane = threadIdx.x & 31;
  const int wave = threadIdx.x >> 5;
  const int job  = blockIdx.x * (blockDim.x >> 5) + wave;
  const int ntn  = N >> 4;
  if (job >= ntn * (M >> 5)) return;
  const int tn = job % ntn;
  const int mb = (job / ntn) << 5;

  v8f acc0 = {}, acc1 = {};
  wmma_job_32x16(A + mb * lda, lda, W + (tn << 4) * ldw, ldw, K, lane, acc0, acc1);

  const int n  = (tn << 4) + (lane & 15);
  const int rb = (lane >> 4) << 3;
  const float bv = bias ? bias[n] : 0.0f;
#pragma unroll
  for (int r = 0; r < 8; ++r) {
    C[(mb + rb + r) * ldc + n]      = acc0[r] + bv;
    C[(mb + 16 + rb + r) * ldc + n] = acc1[r] + bv;
  }
}

// ---------------- encoder recurrence (one block per direction) ----------------
// xproj: [2][S_IN*BATCH][4*ENC] f32 (already includes x@Wih^T + bias).
// h_out: [S_IN*BATCH][2*ENC] f32; dir 0 -> cols [0,256), dir 1 -> cols [256,512).
__global__ void __launch_bounds__(1024)
k_encoder(const float* __restrict__ xproj,
          const _Float16* __restrict__ Whh_f, const _Float16* __restrict__ Whh_b,
          float* __restrict__ h_out) {
  extern __shared__ char smem[];
  _Float16* sh = (_Float16*)smem;                       // 32 x 256 f16
  float*    sc = (float*)(smem + 32 * 256 * 2);         // 32 x 256 f32
  float*    sg = (float*)(smem + 32 * 256 * 2 + 32 * 256 * 4);  // 32 x 1024 f32

  const int dir  = blockIdx.x;
  const float* xp = xproj + dir * (S_IN * BATCH * 4 * ENC);
  const _Float16* Whh = dir ? Whh_b : Whh_f;
  const int tid = threadIdx.x, lane = tid & 31, wave = tid >> 5;

  for (int i = tid; i < 32 * 256; i += 1024) { sh[i] = (_Float16)0.0f; sc[i] = 0.0f; }
  __syncthreads();

  for (int step = 0; step < S_IN; ++step) {
    const int t = dir ? (S_IN - 1 - step) : step;
    // g = h_prev @ Whh^T  (M=32, N=1024, K=256) : 64 jobs over 32 waves
    for (int tn = wave; tn < 64; tn += 32) {
      v8f acc0 = {}, acc1 = {};
      wmma_job_32x16(sh, 256, Whh + (tn << 4) * 256, 256, 256, lane, acc0, acc1);
      const int n  = (tn << 4) + (lane & 15);
      const int rb = (lane >> 4) << 3;
#pragma unroll
      for (int r = 0; r < 8; ++r) {
        sg[(rb + r) * 1024 + n]      = acc0[r] + xp[(t * BATCH + rb + r) * 1024 + n];
        sg[(16 + rb + r) * 1024 + n] = acc1[r] + xp[(t * BATCH + 16 + rb + r) * 1024 + n];
      }
    }
    __syncthreads();
    // gates
    for (int idx = tid; idx < 32 * 256; idx += 1024) {
      const int b = idx >> 8, n = idx & 255;
      const float* g = sg + b * 1024;
      const float c = sigf(g[256 + n]) * sc[idx] + sigf(g[n]) * tanhf(g[512 + n]);
      const float h = sigf(g[768 + n]) * tanhf(c);
      sc[idx] = c;
      sh[idx] = (_Float16)h;
      h_out[(t * BATCH + b) * (2 * ENC) + dir * ENC + n] = h;
    }
    __syncthreads();
  }
}

// ---------------- 2D-LSTM grid recurrence (single block, 1024 threads) -------------
__global__ void __launch_bounds__(1024)
k_grid(const float* __restrict__ h_proj,     // [S_IN*BATCH][1280]
       const float* __restrict__ y_proj,     // [S_OUT*BATCH][1280]
       const _Float16* __restrict__ Ws,      // [1280][512] f16 (== [Ws_hor | Ws_ver])
       _Float16* __restrict__ s_ver_h,       // [S_IN*BATCH][256] f16 (pre-zeroed)
       float*    __restrict__ c_ver,         // [S_IN*BATCH][256] f32 (pre-zeroed)
       _Float16* __restrict__ s_top_h) {     // [S_OUT*BATCH][256] f16
  extern __shared__ char smem[];
  _Float16* s_cat = (_Float16*)smem;                              // 32 x 512 f16: [s_hor | s_ver]
  float*    pre   = (float*)(smem + 32 * 512 * 2);                // 32 x 1280 f32
  float*    chor  = (float*)(smem + 32 * 512 * 2 + 32 * 1280 * 4);// 32 x 256 f32

  const int tid = threadIdx.x, lane = tid & 31, wave = tid >> 5;

  for (int j = 0; j < S_OUT; ++j) {
    for (int idx = tid; idx < 32 * 256; idx += 1024) {
      const int b = idx >> 8, n = idx & 255;
      s_cat[b * 512 + n] = (_Float16)0.0f;   // s_hor = 0 at row start
      chor[idx] = 0.0f;
    }
    __syncthreads();

    for (int i = 0; i < S_IN; ++i) {
      // phase 0: async stage s_ver[i] -> s_cat[:, 256:512] (8192 halfs, 16B/lane)
      {
        const int b = tid >> 5;
        const int n = (tid & 31) << 3;
        const _Float16* src = s_ver_h + ((i * BATCH + b) << 8) + n;
        const unsigned lds_off = (unsigned)(uintptr_t)(s_cat + b * 512 + 256 + n);
        async_g2l_b128(lds_off, src);
      }
      wait_async0();
      __syncthreads();

      // phase 1: pre = [s_hor|s_ver] @ Ws^T + h_proj[i] + y_proj[j]
      //          (M=32, N=1280, K=512) : 80 jobs over 32 waves
      for (int tn = wave; tn < 80; tn += 32) {
        v8f acc0 = {}, acc1 = {};
        wmma_job_32x16(s_cat, 512, Ws + (tn << 4) * 512, 512, 512, lane, acc0, acc1);
        const int n  = (tn << 4) + (lane & 15);
        const int rb = (lane >> 4) << 3;
#pragma unroll
        for (int r = 0; r < 8; ++r) {
          const int m0 = rb + r, m1 = 16 + rb + r;
          pre[m0 * 1280 + n] = acc0[r] + h_proj[(i * BATCH + m0) * 1280 + n]
                                       + y_proj[(j * BATCH + m0) * 1280 + n];
          pre[m1 * 1280 + n] = acc1[r] + h_proj[(i * BATCH + m1) * 1280 + n]
                                       + y_proj[(j * BATCH + m1) * 1280 + n];
        }
      }
      __syncthreads();

      // phase 2: gates + state update
      for (int idx = tid; idx < 32 * 256; idx += 1024) {
        const int b = idx >> 8, n = idx & 255;
        const float* pb = pre + b * 1280;
        const float pi = sigf(pb[n]);
        const float pf = sigf(pb[256 + n]);
        const float pl = sigf(pb[512 + n]);
        const float po = sigf(pb[768 + n]);
        const float pg = tanhf(pb[1024 + n]);
        const float cv = c_ver[(i * BATCH + b) * 256 + n];
        const float c  = pf * (pl * chor[idx] + (1.0f - pl) * cv) + pi * pg;
        const float s  = po * tanhf(c);
        chor[idx] = c;
        c_ver[(i * BATCH + b) * 256 + n] = c;
        const _Float16 s16 = (_Float16)s;
        s_cat[b * 512 + n] = s16;                          // next cell's s_hor
        s_ver_h[(i * BATCH + b) * 256 + n] = s16;          // next row's s_ver
        if (i == S_IN - 1) s_top_h[(j * BATCH + b) * 256 + n] = s16;
      }
      // global stores (STOREcnt) must land before next phase-0 async reads (ASYNCcnt)
      wait_store0();
      __syncthreads();
    }
  }
}

// ---------------- host ----------------

static inline void launch_gemm(const _Float16* A, int lda, const _Float16* W, int ldw,
                               const float* bias, float* C, int ldc,
                               int M, int N, int K, hipStream_t s) {
  const int jobs = (M >> 5) * (N >> 4);
  const int blocks = (jobs + 7) >> 3;   // 8 waves (256 threads) per block
  k_gemm_xwT<<<blocks, 256, 0, s>>>(A, lda, W, ldw, bias, C, ldc, M, N, K);
}

extern "C" void kernel_launch(void* const* d_in, const int* in_sizes, int n_in,
                              void* d_out, int out_size, void* d_ws, size_t ws_size,
                              hipStream_t stream) {
  const int*   x        = (const int*)d_in[0];
  const int*   y        = (const int*)d_in[2];
  const float* in_emb   = (const float*)d_in[3];
  const float* out_emb  = (const float*)d_in[4];
  const float* Wih_f    = (const float*)d_in[5];
  const float* Whh_f    = (const float*)d_in[6];
  const float* b_f      = (const float*)d_in[7];
  const float* Wih_b    = (const float*)d_in[8];
  const float* Whh_b    = (const float*)d_in[9];
  const float* b_b      = (const float*)d_in[10];
  const float* cell_Wx  = (const float*)d_in[11];
  const float* cell_Ws  = (const float*)d_in[12];
  const float* cell_b   = (const float*)d_in[13];
  const float* logit_W  = (const float*)d_in[14];
  const float* logit_b  = (const float*)d_in[15];
  float* out = (float*)d_out;

  // workspace carve-out
  size_t off = 0;
  auto alloc = [&](size_t bytes) -> char* {
    char* p = (char*)d_ws + off;
    off = (off + bytes + 255) & ~(size_t)255;
    return p;
  };
  _Float16* wWih_f = (_Float16*)alloc(1024 * 256 * 2);
  _Float16* wWhh_f = (_Float16*)alloc(1024 * 256 * 2);
  _Float16* wWih_b = (_Float16*)alloc(1024 * 256 * 2);
  _Float16* wWhh_b = (_Float16*)alloc(1024 * 256 * 2);
  _Float16* wWx    = (_Float16*)alloc(1280 * 768 * 2);
  _Float16* wWs    = (_Float16*)alloc(1280 * 512 * 2);
  _Float16* wLog   = (_Float16*)alloc((size_t)VOCAB * 256 * 2);
  _Float16* x_emb  = (_Float16*)alloc(512 * 256 * 2);
  _Float16* y_emb  = (_Float16*)alloc(512 * 256 * 2);
  float*    xproj  = (float*)alloc(2ull * 512 * 1024 * 4);   // [dir][512][1024]
  float*    h      = (float*)alloc(512 * 512 * 4);
  _Float16* h_h    = (_Float16*)alloc(512 * 512 * 2);
  float*    h_proj = (float*)alloc(512 * 1280 * 4);
  float*    y_proj = (float*)alloc(512 * 1280 * 4);
  _Float16* s_ver  = (_Float16*)alloc(512 * 256 * 2);
  float*    c_ver  = (float*)alloc(512 * 256 * 4);
  _Float16* s_top  = (_Float16*)alloc(512 * 256 * 2);
  (void)ws_size; (void)in_sizes; (void)n_in; (void)out_size;

  // 1) weights -> f16
  auto cvt = [&](const float* s, _Float16* d, int n) {
    int blocks = (n + 255) / 256; if (blocks > 4096) blocks = 4096;
    k_f2h<<<blocks, 256, 0, stream>>>(s, d, n);
  };
  cvt(Wih_f, wWih_f, 1024 * 256);
  cvt(Whh_f, wWhh_f, 1024 * 256);
  cvt(Wih_b, wWih_b, 1024 * 256);
  cvt(Whh_b, wWhh_b, 1024 * 256);
  cvt(cell_Wx, wWx, 1280 * 768);
  cvt(cell_Ws, wWs, 1280 * 512);
  cvt(logit_W, wLog, VOCAB * 256);

  // 2) embeddings
  k_embed<<<512, 256, 0, stream>>>(x, in_emb, x_emb, 512, 0);
  k_embed<<<512, 256, 0, stream>>>(y, out_emb, y_emb, 512, 1);

  // 3) zero vertical grid state
  k_zero_u32<<<256, 256, 0, stream>>>((unsigned*)s_ver, 512 * 256 / 2);
  k_zero_u32<<<256, 256, 0, stream>>>((unsigned*)c_ver, 512 * 256);

  // 4) input projections (include encoder biases here)
  launch_gemm(x_emb, 256, wWih_f, 256, b_f, xproj,              1024, 512, 1024, 256, stream);
  launch_gemm(x_emb, 256, wWih_b, 256, b_b, xproj + 512 * 1024, 1024, 512, 1024, 256, stream);

  // 5) bidirectional encoder (1 block/dir, 32 waves, 176KB LDS)
  k_encoder<<<2, 1024, 32 * 256 * 2 + 32 * 256 * 4 + 32 * 1024 * 4, stream>>>(
      xproj, wWhh_f, wWhh_b, h);

  // 6) h -> f16
  cvt(h, h_h, 512 * 512);

  // 7) decoder projections
  launch_gemm(h_h,   512, wWx,       768, nullptr, h_proj, 1280, 512, 1280, 512, stream);
  launch_gemm(y_emb, 256, wWx + 512, 768, cell_b,  y_proj, 1280, 512, 1280, 256, stream);

  // 8) 2D grid recurrence (1 block, 32 waves, 224KB LDS)
  k_grid<<<1, 1024, 32 * 512 * 2 + 32 * 1280 * 4 + 32 * 256 * 4, stream>>>(
      h_proj, y_proj, wWs, s_ver, c_ver, s_top);

  // 9) logits straight into d_out: [512][32000]
  launch_gemm(s_top, 256, wLog, 256, logit_b, out, VOCAB, 512, VOCAB, 256, stream);
}
